// StateEncoderCompact_18305150615560
// MI455X (gfx1250) — matrix operand order repaired and checked by reading, not compiled, not run on hardware
//
#include <hip/hip_runtime.h>
#include <hip/hip_bf16.h>

// ---------------------------------------------------------------------------
// StateEncoderCompact for MI455X (gfx1250): wave32 + v_wmma_f32_16x16x32_f16.
// All WMMA operands are unconditional 16B-aligned loads (no exec churn):
// weights pre-padded/converted to f16 in d_ws, activations staged in padded
// f16 LDS, K-padding handled by zero blocks instead of per-element branches.
// History tile staged via gfx1250 async global->LDS DMA (ASYNCcnt).
// ---------------------------------------------------------------------------
#define BATCH    8192
#define DOBS     2988      // 165 + 61*40 + 383
#define HIST_OFF 165
#define STAT_OFF 2605      // 165 + 61*40

typedef __attribute__((ext_vector_type(16))) _Float16     v16h;
typedef __attribute__((ext_vector_type(8)))  float        v8f;
typedef __attribute__((ext_vector_type(4)))  unsigned int u32x4;

union V16 { v16h h; u32x4 q[2]; };

struct P {
  const float *obs, *seat, *card_emb, *w_cnt_hand, *b_cnt_hand, *w_cnt_cap, *b_cnt_cap,
              *h2t_win, *h2t_bin, *h2t_wout, *h2t_bout,
              *t2h_win, *t2h_bin, *t2h_wout, *t2h_bout,
              *w_merge, *b_merge, *w_hproj, *b_hproj, *pos_emb,
              *enc_win, *enc_bin, *enc_wout, *enc_bout,
              *ln1_g, *ln1_b, *w_ff1, *b_ff1, *w_ff2, *b_ff2, *ln2_g, *ln2_b,
              *w_stats, *b_stats, *w_seat, *b_seat, *w_comb, *b_comb;
  float *ws_hist, *ws_set, *out;
  // pre-converted f16 weights (padded K), in d_ws
  _Float16 *hp, *ewin, *ewout, *ff1, *ff2, *cwin[2], *cwout[2], *mrg, *stt, *cmb;
};

__device__ inline v8f wmma16(v16h a, v16h b, v8f c) {
  return __builtin_amdgcn_wmma_f32_16x16x32_f16(false, a, false, b, (short)0, c,
                                                false, false);
}

// ---- fragment loaders: contiguous-K striping (same for A and B), pure b128 loads --------------------
// A 16x32 f16: lane element j = K grp*16+j, row m = lane&15.
__device__ inline v16h ld_a(const _Float16* lo, int stride) {
  const int lane = threadIdx.x & 31;
  const int row = lane & 15, grp = lane >> 4;
  const _Float16* bp = lo + row * stride + grp * 16;
  V16 fr;
  fr.q[0] = *(const u32x4*)(bp);
  fr.q[1] = *(const u32x4*)(bp + 8);
  return fr.h;
}
// A with independent source for the upper 16 K (pass lo==hi to read garbage
// that a zeroed B half will annihilate).
__device__ inline v16h ld_a2(const _Float16* lo, const _Float16* hi, int stride) {
  const int lane = threadIdx.x & 31;
  const int row = lane & 15, grp = lane >> 4;
  const _Float16* sp = grp ? (hi + row * stride) : (lo + row * stride);
  V16 fr;
  fr.q[0] = *(const u32x4*)(sp);
  fr.q[1] = *(const u32x4*)(sp + 8);
  return fr.h;
}
// B 32x16 f16: lane element j = K grp*16+j, column n = lane&15.
__device__ inline v16h ld_b(const _Float16* base, int stride) {
  const int lane = threadIdx.x & 31;
  const int n = lane & 15, grp = lane >> 4;
  const _Float16* sp = base + n * stride + grp * 16;
  V16 fr;
  fr.q[0] = *(const u32x4*)(sp);
  fr.q[1] = *(const u32x4*)(sp + 8);
  return fr.h;
}
// B with separate source/stride for K 16..31 (use a 16-half zero block,
// stride 0, to zero the padded K half without branches).
__device__ inline v16h ld_b2(const _Float16* lo, int sLo, const _Float16* hi, int sHi) {
  const int lane = threadIdx.x & 31;
  const int n = lane & 15, grp = lane >> 4;
  const _Float16* sp = grp ? (hi + n * sHi) : (lo + n * sLo);
  V16 fr;
  fr.q[0] = *(const u32x4*)(sp);
  fr.q[1] = *(const u32x4*)(sp + 8);
  return fr.h;
}
// C/D 16x16 f32: VGPR i -> M = i + 8*(lane/16), N = lane%16.
template <typename F>
__device__ inline void store_c(v8f c, F put) {
  const int lane = threadIdx.x & 31;
  const int n = lane & 15, grp = lane >> 4;
#pragma unroll
  for (int i = 0; i < 8; ++i) put(grp * 8 + i, n, c[i]);
}

// ---------------------------------------------------------------------------
// Prep: convert all GEMM weights to f16 with K padded (zeros) so B fragments
// are unconditional.  W row-major (N,K) -> dst (N,Kpad).
// ---------------------------------------------------------------------------
__device__ inline void cvtpad(_Float16* dst, const float* src, int N, int K, int Kp,
                              int g, int gs) {
  for (int i = g; i < N * Kp; i += gs) {
    int n = i / Kp, k = i - n * Kp;
    dst[i] = (_Float16)((k < K) ? src[n * K + k] : 0.f);
  }
}
__global__ __launch_bounds__(256) void prep_kernel(P p) {
  int g = blockIdx.x * 256 + threadIdx.x, gs = gridDim.x * 256;
  cvtpad(p.hp,       p.w_hproj,  64, 61, 64, g, gs);
  cvtpad(p.ewin,     p.enc_win, 192, 64, 64, g, gs);
  cvtpad(p.ewout,    p.enc_wout, 64, 64, 64, g, gs);
  cvtpad(p.ff1,      p.w_ff1,  128, 64, 64, g, gs);
  cvtpad(p.ff2,      p.w_ff2,   64, 128, 128, g, gs);
  cvtpad(p.cwin[0],  p.h2t_win,  96, 32, 32, g, gs);
  cvtpad(p.cwout[0], p.h2t_wout, 32, 32, 32, g, gs);
  cvtpad(p.cwin[1],  p.t2h_win,  96, 32, 32, g, gs);
  cvtpad(p.cwout[1], p.t2h_wout, 32, 32, 32, g, gs);
  cvtpad(p.mrg,      p.w_merge,  64, 224, 224, g, gs);
  cvtpad(p.stt,      p.w_stats,  64, 383, 384, g, gs);
  cvtpad(p.cmb,      p.w_comb,  256, 224, 224, g, gs);
}

// ---------------------------------------------------------------------------
// Kernel 1: history transformer branch; one block (8 waves) per element.
// ---------------------------------------------------------------------------
__global__ __launch_bounds__(256) void hist_kernel(P p) {
  const int b = blockIdx.x, tid = threadIdx.x, wave = tid >> 5;
  __shared__ __align__(16) float    s_x[48 * 64];      // raw hist -> f32 residual
  __shared__ __align__(16) _Float16 s_bufA[48 * 64];   // hist f16 -> LN1 f16
  __shared__ __align__(16) _Float16 s_bufB[48 * 64];   // hseq f16 -> attn out
  __shared__ __align__(16) _Float16 s_q[48 * 64], s_k[48 * 64];
  __shared__ __align__(16) _Float16 s_vT[64 * 64];     // V transposed [dim][seq]
  __shared__ __align__(16) float    s_s[48 * 64];      // scores; alias f16 FFN hidden
  __shared__ __align__(16) _Float16 s_p[48 * 64];      // probs (cols>=40 zeroed)
  __shared__ __align__(16) _Float16 s_zero[16];
  _Float16* s_h = (_Float16*)s_s;                      // 48x128 f16

  if (tid < 16) s_zero[tid] = (_Float16)0.f;

  // ---- stage raw f32 history into s_x via async global->LDS DMA (ASYNCcnt),
  //      zero the pads, then convert to f16 into s_bufA ----
  {
    const unsigned ldsbase = (unsigned)(unsigned long long)(void*)s_x;  // LDS byte offset
    const unsigned gbase = (unsigned)(((size_t)b * DOBS + HIST_OFF) * 4u);
    for (int i = tid; i < 40 * 61; i += 256) {
      int r = i / 61, kk = i - r * 61;
      unsigned lds = ldsbase + (unsigned)(r * 64 + kk) * 4u;
      unsigned gof = gbase + (unsigned)i * 4u;
      asm volatile("global_load_async_to_lds_b32 %0, %1, %2"
                   :: "v"(lds), "v"(gof), "s"(p.obs)
                   : "memory");
    }
    for (int i = tid; i < 48 * 64; i += 256) {           // zero pad region
      int r = i >> 6, kk = i & 63;
      if (r >= 40 || kk >= 61) s_x[i] = 0.f;
    }
    asm volatile("s_wait_asynccnt 0x0" ::: "memory");
    __syncthreads();
    for (int i = tid; i < 48 * 64; i += 256)
      s_bufA[i] = (_Float16)s_x[i];
  }
  __syncthreads();

  // ---- hseq = hist @ Whp^T + b + pos : M=48 N=64 K=64 ----
  for (int t = wave; t < 12; t += 8) {
    int mt = t >> 2, nt = t & 3;
    v8f acc = {};
#pragma unroll
    for (int ks = 0; ks < 2; ++ks) {
      v16h a  = ld_a(s_bufA + mt * 16 * 64 + ks * 32, 64);
      v16h bm = ld_b(p.hp + nt * 16 * 64 + ks * 32, 64);
      acc = wmma16(a, bm, acc);
    }
    store_c(acc, [&](int m, int n, float v) {
      int row = mt * 16 + m, col = nt * 16 + n;
      int rc = row < 40 ? row : 39;                    // clamped, branch-free
      float pe = p.pos_emb[rc * 64 + col];
      float r = v + p.b_hproj[col] + (row < 40 ? pe : 0.f);
      s_x[row * 64 + col] = r;
      s_bufB[row * 64 + col] = (_Float16)r;
    });
  }
  __syncthreads();

  // ---- in_proj: q|k|vT = hseq @ enc_win^T + enc_bin (N=192) ----
  for (int t = wave; t < 36; t += 8) {
    int mt = t / 12, nt = t % 12;
    v8f acc = {};
#pragma unroll
    for (int ks = 0; ks < 2; ++ks) {
      v16h a  = ld_a(s_bufB + mt * 16 * 64 + ks * 32, 64);
      v16h bm = ld_b(p.ewin + nt * 16 * 64 + ks * 32, 64);
      acc = wmma16(a, bm, acc);
    }
    store_c(acc, [&](int m, int n, float v) {
      int row = mt * 16 + m, col = nt * 16 + n;       // col block uniform per tile
      float r = v + p.enc_bin[col];
      if (col < 64)       s_q[row * 64 + col] = (_Float16)r;
      else if (col < 128) s_k[row * 64 + col - 64] = (_Float16)r;
      else                s_vT[(col - 128) * 64 + row] = (_Float16)r;
    });
  }
  __syncthreads();

  // ---- attention: 4 heads of dim 16 ----
  for (int h = 0; h < 4; ++h) {
    const int d0 = h * 16;
    for (int t = wave; t < 9; t += 8) {               // scores QK^T (K=16 -> 32)
      int mt = t / 3, nt = t % 3;
      const _Float16* alo = s_q + mt * 16 * 64 + d0;
      v16h a  = ld_a2(alo, alo, 64);                  // upper K garbage
      v16h bm = ld_b2(s_k + nt * 16 * 64 + d0, 64, s_zero, 0);  // upper K zero
      v8f acc = {};
      acc = wmma16(a, bm, acc);
      store_c(acc, [&](int m, int n, float v) {
        s_s[(mt * 16 + m) * 64 + nt * 16 + n] = v * 0.25f;      // 1/sqrt(16)
      });
    }
    __syncthreads();
    if (tid < 48) {                                    // f32 softmax per row
      float mx = -1e30f;
      for (int j = 0; j < 40; ++j) mx = fmaxf(mx, s_s[tid * 64 + j]);
      float sum = 0.f;
      for (int j = 0; j < 40; ++j) {
        float e = __expf(s_s[tid * 64 + j] - mx);
        s_s[tid * 64 + j] = e;
        sum += e;
      }
      float inv = 1.f / sum;
      for (int j = 0; j < 40; ++j)
        s_p[tid * 64 + j] = (_Float16)(s_s[tid * 64 + j] * inv);
      for (int j = 40; j < 64; ++j) s_p[tid * 64 + j] = (_Float16)0.f;
    }
    __syncthreads();
    for (int t = wave; t < 3; t += 8) {                // o = P @ V (K=64)
      int mt = t;
      v8f acc = {};
#pragma unroll
      for (int ks = 0; ks < 2; ++ks) {
        v16h a  = ld_a(s_p + mt * 16 * 64 + ks * 32, 64);
        v16h bm = ld_b(s_vT + d0 * 64 + ks * 32, 64);  // pad seq killed by P zeros
        acc = wmma16(a, bm, acc);
      }
      store_c(acc, [&](int m, int n, float v) {
        s_bufB[(mt * 16 + m) * 64 + d0 + n] = (_Float16)v;
      });
    }
    __syncthreads();
  }

  // ---- out_proj + residual ----
  for (int t = wave; t < 12; t += 8) {
    int mt = t >> 2, nt = t & 3;
    v8f acc = {};
#pragma unroll
    for (int ks = 0; ks < 2; ++ks) {
      v16h a  = ld_a(s_bufB + mt * 16 * 64 + ks * 32, 64);
      v16h bm = ld_b(p.ewout + nt * 16 * 64 + ks * 32, 64);
      acc = wmma16(a, bm, acc);
    }
    store_c(acc, [&](int m, int n, float v) {
      int row = mt * 16 + m, col = nt * 16 + n;
      s_x[row * 64 + col] += v + p.enc_bout[col];
    });
  }
  __syncthreads();

  // ---- LayerNorm 1 (f32) + f16 copy for FFN A ----
  if (tid < 48) {
    float mean = 0.f;
    for (int j = 0; j < 64; ++j) mean += s_x[tid * 64 + j];
    mean *= (1.f / 64.f);
    float var = 0.f;
    for (int j = 0; j < 64; ++j) { float d = s_x[tid * 64 + j] - mean; var += d * d; }
    float inv = rsqrtf(var * (1.f / 64.f) + 1e-5f);
    for (int j = 0; j < 64; ++j) {
      float xn = (s_x[tid * 64 + j] - mean) * inv * p.ln1_g[j] + p.ln1_b[j];
      s_x[tid * 64 + j] = xn;
      s_bufA[tid * 64 + j] = (_Float16)xn;
    }
  }
  __syncthreads();

  // ---- FFN1: relu(x @ w_ff1^T + b) N=128 ----
  for (int t = wave; t < 24; t += 8) {
    int mt = t >> 3, nt = t & 7;
    v8f acc = {};
#pragma unroll
    for (int ks = 0; ks < 2; ++ks) {
      v16h a  = ld_a(s_bufA + mt * 16 * 64 + ks * 32, 64);
      v16h bm = ld_b(p.ff1 + nt * 16 * 64 + ks * 32, 64);
      acc = wmma16(a, bm, acc);
    }
    store_c(acc, [&](int m, int n, float v) {
      int row = mt * 16 + m, col = nt * 16 + n;
      s_h[row * 128 + col] = (_Float16)fmaxf(v + p.b_ff1[col], 0.f);
    });
  }
  __syncthreads();

  // ---- FFN2 + residual ----
  for (int t = wave; t < 12; t += 8) {
    int mt = t >> 2, nt = t & 3;
    v8f acc = {};
#pragma unroll
    for (int ks = 0; ks < 4; ++ks) {
      v16h a  = ld_a(s_h + mt * 16 * 128 + ks * 32, 128);
      v16h bm = ld_b(p.ff2 + nt * 16 * 128 + ks * 32, 128);
      acc = wmma16(a, bm, acc);
    }
    store_c(acc, [&](int m, int n, float v) {
      int row = mt * 16 + m, col = nt * 16 + n;
      s_x[row * 64 + col] = v + p.b_ff2[col] + s_x[row * 64 + col];
    });
  }
  __syncthreads();

  // ---- LayerNorm 2 + mean over 40 rows ----
  if (tid < 48) {
    float mean = 0.f;
    for (int j = 0; j < 64; ++j) mean += s_x[tid * 64 + j];
    mean *= (1.f / 64.f);
    float var = 0.f;
    for (int j = 0; j < 64; ++j) { float d = s_x[tid * 64 + j] - mean; var += d * d; }
    float inv = rsqrtf(var * (1.f / 64.f) + 1e-5f);
    for (int j = 0; j < 64; ++j)
      s_x[tid * 64 + j] = (s_x[tid * 64 + j] - mean) * inv * p.ln2_g[j] + p.ln2_b[j];
  }
  __syncthreads();
  if (tid < 64) {
    float s = 0.f;
    for (int r = 0; r < 40; ++r) s += s_x[r * 64 + tid];
    p.ws_hist[(size_t)b * 64 + tid] = s * (1.f / 40.f);
  }
}

// ---------------------------------------------------------------------------
// Kernel 2: set branch (two masked 40x32 cross-attentions + small features).
// ---------------------------------------------------------------------------
__global__ __launch_bounds__(256) void set_kernel(P p) {
  const int b = blockIdx.x, tid = threadIdx.x, wave = tid >> 5;
  __shared__ __align__(16) _Float16 s_hand[48 * 32], s_table[48 * 32];
  __shared__ __align__(16) _Float16 s_q[48 * 64], s_k[48 * 64];  // per-head slots of 16 (8 valid + 8 zero)
  __shared__ __align__(16) _Float16 s_vT[48 * 64];               // [dim(pad48)][seq64]
  __shared__ __align__(16) float    s_s[48 * 64];
  __shared__ __align__(16) _Float16 s_p[48 * 64];
  __shared__ __align__(16) _Float16 s_ao[48 * 32];
  __shared__ __align__(16) float    s_att[48 * 32];
  __shared__ __align__(16) _Float16 s_zero[16];
  __shared__ float s_mask[2][40];
  __shared__ int   s_cnt[2];
  __shared__ float s_feat[224];

  const float* ob = p.obs + (size_t)b * DOBS;
  if (tid < 16) s_zero[tid] = (_Float16)0.f;
  for (int i = tid; i < 48 * 64; i += 256) s_k[i] = (_Float16)0.f;  // zero pad slots
  if (tid < 40)               s_mask[0][tid]      = ob[tid];
  if (tid >= 64 && tid < 104) s_mask[1][tid - 64] = ob[43 + tid - 64];
  __syncthreads();
  for (int i = tid; i < 48 * 32; i += 256) {
    int r = i >> 5, c = i & 31;
    float e = (r < 40) ? p.card_emb[r * 32 + c] : 0.f;
    s_hand[i]  = (_Float16)((r < 40 ? s_mask[0][r] : 0.f) * e);
    s_table[i] = (_Float16)((r < 40 ? s_mask[1][r] : 0.f) * e);
  }
  if (tid < 2) {
    int c = 0;
    for (int j = 0; j < 40; ++j) c += (s_mask[tid][j] > 0.5f) ? 1 : 0;
    s_cnt[tid] = c;
  }
  __syncthreads();

  for (int pass = 0; pass < 2; ++pass) {    // 0: h2t (Q=hand,KV=table), 1: t2h
    const _Float16* Qs   = pass ? s_table : s_hand;
    const _Float16* Ks   = pass ? s_hand  : s_table;
    const _Float16* win  = p.cwin[pass];
    const _Float16* wout = p.cwout[pass];
    const float* bin  = pass ? p.t2h_bin  : p.h2t_bin;
    const float* bout = pass ? p.t2h_bout : p.h2t_bout;
    const int keyIdx = pass ? 0 : 1;
    const int qIdx   = pass ? 1 : 0;

    // in_proj (N=96, K=32): q -> head slots, k -> head slots (pads stay 0), v -> vT
    for (int t = wave; t < 18; t += 8) {
      int mt = t / 6, nt = t % 6;
      const _Float16* Asrc = (nt < 2) ? Qs : Ks;
      v16h a  = ld_a(Asrc + mt * 16 * 32, 32);
      v16h bm = ld_b(win + nt * 16 * 32, 32);
      v8f acc = {};
      acc = wmma16(a, bm, acc);
      store_c(acc, [&](int m, int n, float v) {
        int row = mt * 16 + m, col = nt * 16 + n;
        float r = v + bin[col];
        if (col < 32) {
          int hh = col >> 3, w = col & 7;
          s_q[row * 64 + hh * 16 + w] = (_Float16)r;
        } else if (col < 64) {
          int c2 = col - 32, hh = c2 >> 3, w = c2 & 7;
          s_k[row * 64 + hh * 16 + w] = (_Float16)r;
        } else {
          s_vT[(col - 64) * 64 + row] = (_Float16)r;
        }
      });
    }
    __syncthreads();

    for (int h = 0; h < 4; ++h) {           // head dim 8
      for (int t = wave; t < 9; t += 8) {   // scores (K=8 valid; 8 zeros; 16 zeros)
        int mt = t / 3, nt = t % 3;
        const _Float16* alo = s_q + mt * 16 * 64 + h * 16;
        v16h a  = ld_a2(alo, alo, 64);
        v16h bm = ld_b2(s_k + nt * 16 * 64 + h * 16, 64, s_zero, 0);
        v8f acc = {};
        acc = wmma16(a, bm, acc);
        store_c(acc, [&](int m, int n, float v) {
          s_s[(mt * 16 + m) * 64 + nt * 16 + n] = v * 0.35355339f;  // 1/sqrt(8)
        });
      }
      __syncthreads();
      if (tid < 48) {                        // masked softmax
        const float* km = s_mask[keyIdx];
        float mx = -1e30f;
        for (int j = 0; j < 40; ++j)
          if (km[j] > 0.5f) mx = fmaxf(mx, s_s[tid * 64 + j]);
        float sum = 0.f;
        for (int j = 0; j < 40; ++j) {
          float e = (km[j] > 0.5f) ? __expf(s_s[tid * 64 + j] - mx) : 0.f;
          s_s[tid * 64 + j] = e;
          sum += e;
        }
        float inv = sum > 0.f ? 1.f / sum : 0.f;
        for (int j = 0; j < 40; ++j)
          s_p[tid * 64 + j] = (_Float16)(s_s[tid * 64 + j] * inv);
        for (int j = 40; j < 64; ++j) s_p[tid * 64 + j] = (_Float16)0.f;
      }
      __syncthreads();
      for (int t = wave; t < 3; t += 8) {    // o = P @ V
        int mt = t;
        v8f acc = {};
#pragma unroll
        for (int ks = 0; ks < 2; ++ks) {
          v16h a  = ld_a(s_p + mt * 16 * 64 + ks * 32, 64);
          v16h bm = ld_b(s_vT + (h * 8) * 64 + ks * 32, 64);
          acc = wmma16(a, bm, acc);
        }
        store_c(acc, [&](int m, int n, float v) {
          if (n < 8) s_ao[(mt * 16 + m) * 32 + h * 8 + n] = (_Float16)v;
        });
      }
      __syncthreads();
    }

    // out_proj (32x32)
    for (int t = wave; t < 6; t += 8) {
      int mt = t >> 1, nt = t & 1;
      v16h a  = ld_a(s_ao + mt * 16 * 32, 32);
      v16h bm = ld_b(wout + nt * 16 * 32, 32);
      v8f acc = {};
      acc = wmma16(a, bm, acc);
      store_c(acc, [&](int m, int n, float v) {
        s_att[(mt * 16 + m) * 32 + nt * 16 + n] = v + bout[nt * 16 + n];
      });
    }
    __syncthreads();

    if (tid < 32) {       // masked mean over query rows; zero if no keys present
      const float* qm = s_mask[qIdx];
      float s = 0.f;
      int c = 0;
      for (int r = 0; r < 40; ++r)
        if (qm[r] > 0.5f) { s += s_att[r * 32 + tid]; ++c; }
      float v = (s_cnt[keyIdx] > 0) ? s / fmaxf((float)c, 1.f) : 0.f;
      s_feat[128 + pass * 32 + tid] = v;
    }
    __syncthreads();
  }

  // embedding sums + count MLPs
  for (int i = tid; i < 128; i += 256) {
    int f = i >> 5, c = i & 31;
    const float* m = (f == 0) ? ob : (f == 1) ? ob + 43 : (f == 2) ? ob + 83
                                                                   : ob + 123;
    float s = 0.f;
    for (int r = 0; r < 40; ++r) s += m[r] * p.card_emb[r * 32 + c];
    s_feat[f * 32 + c] = s;
  }
  if (tid >= 128 && tid < 144) {
    int q = tid - 128;
    float s = p.b_cnt_hand[q];
    for (int j = 0; j < 3; ++j) s += ob[40 + j] * p.w_cnt_hand[q * 3 + j];
    s_feat[192 + q] = fmaxf(s, 0.f);
  }
  if (tid >= 144 && tid < 160) {
    int q = tid - 144;
    float s = p.b_cnt_cap[q];
    for (int j = 0; j < 2; ++j) s += ob[163 + j] * p.w_cnt_cap[q * 2 + j];
    s_feat[208 + q] = fmaxf(s, 0.f);
  }
  __syncthreads();
  if (tid < 224) p.ws_set[(size_t)b * 224 + tid] = s_feat[tid];
}

// ---------------------------------------------------------------------------
// Kernel 3: batched tail GEMMs (batch tile of 16 = WMMA M).
// ---------------------------------------------------------------------------
__global__ __launch_bounds__(256) void tail_kernel(P p) {
  const int e0 = blockIdx.x * 16, tid = threadIdx.x, wave = tid >> 5;
  __shared__ __align__(16) _Float16 s_stat[16 * 384];
  __shared__ __align__(16) _Float16 s_set[16 * 224];
  __shared__ __align__(16) _Float16 s_c[16 * 224];

  for (int i = tid; i < 16 * 384; i += 256) {
    int r = i / 384, k = i - r * 384;
    s_stat[i] = (_Float16)((k < 383) ? p.obs[(size_t)(e0 + r) * DOBS + STAT_OFF + k] : 0.f);
  }
  for (int i = tid; i < 16 * 224; i += 256) {
    int r = i / 224, k = i - r * 224;
    s_set[i] = (_Float16)p.ws_set[(size_t)(e0 + r) * 224 + k];
  }
  __syncthreads();

  // stats_feat (4 tiles, K=384) and set_feat=merge (4 tiles, K=224)
  for (int t = wave; t < 8; t += 8) {
    if (t < 4) {
      int nt = t;
      v8f acc = {};
      for (int ks = 0; ks < 12; ++ks) {
        v16h a  = ld_a(s_stat + ks * 32, 384);
        v16h bm = ld_b(p.stt + nt * 16 * 384 + ks * 32, 384);
        acc = wmma16(a, bm, acc);
      }
      store_c(acc, [&](int m, int n, float v) {
        int col = nt * 16 + n;
        s_c[m * 224 + 128 + col] = (_Float16)fmaxf(v + p.b_stats[col], 0.f);
      });
    } else {
      int nt = t - 4;
      v8f acc = {};
      for (int ks = 0; ks < 7; ++ks) {
        v16h a  = ld_a(s_set + ks * 32, 224);
        v16h bm = ld_b(p.mrg + nt * 16 * 224 + ks * 32, 224);
        acc = wmma16(a, bm, acc);
      }
      store_c(acc, [&](int m, int n, float v) {
        int col = nt * 16 + n;
        s_c[m * 224 + col] = (_Float16)fmaxf(v + p.b_merge[col], 0.f);
      });
    }
  }
  for (int i = tid; i < 16 * 64; i += 256) {
    int r = i >> 6, c = i & 63;
    s_c[r * 224 + 64 + c] = (_Float16)p.ws_hist[(size_t)(e0 + r) * 64 + c];
  }
  for (int i = tid; i < 16 * 32; i += 256) {
    int r = i >> 5, q = i & 31;
    float s = p.b_seat[q];
    for (int j = 0; j < 6; ++j)
      s += p.seat[(size_t)(e0 + r) * 6 + j] * p.w_seat[q * 6 + j];
    s_c[r * 224 + 192 + q] = (_Float16)fmaxf(s, 0.f);
  }
  __syncthreads();

  // out = relu(combined @ w_comb^T + b_comb)  (N=256, K=224)
  for (int t = wave; t < 16; t += 8) {
    int nt = t;
    v8f acc = {};
    for (int ks = 0; ks < 7; ++ks) {
      v16h a  = ld_a(s_c + ks * 32, 224);
      v16h bm = ld_b(p.cmb + nt * 16 * 224 + ks * 32, 224);
      acc = wmma16(a, bm, acc);
    }
    store_c(acc, [&](int m, int n, float v) {
      int col = nt * 16 + n;
      p.out[(size_t)(e0 + m) * 256 + col] = fmaxf(v + p.b_comb[col], 0.f);
    });
  }
}

// ---------------------------------------------------------------------------
extern "C" void kernel_launch(void* const* d_in, const int* in_sizes, int n_in,
                              void* d_out, int out_size, void* d_ws, size_t ws_size,
                              hipStream_t stream) {
  (void)in_sizes; (void)n_in; (void)out_size; (void)ws_size;
  const float* const* f = (const float* const*)d_in;
  P p;
  p.obs = f[0];  p.seat = f[1];  p.card_emb = f[2];
  p.w_cnt_hand = f[3];  p.b_cnt_hand = f[4];
  p.w_cnt_cap = f[5];   p.b_cnt_cap = f[6];
  p.h2t_win = f[7];  p.h2t_bin = f[8];  p.h2t_wout = f[9];  p.h2t_bout = f[10];
  p.t2h_win = f[11]; p.t2h_bin = f[12]; p.t2h_wout = f[13]; p.t2h_bout = f[14];
  p.w_merge = f[15]; p.b_merge = f[16];
  p.w_hproj = f[17]; p.b_hproj = f[18]; p.pos_emb = f[19];
  p.enc_win = f[20]; p.enc_bin = f[21]; p.enc_wout = f[22]; p.enc_bout = f[23];
  p.ln1_g = f[24]; p.ln1_b = f[25];
  p.w_ff1 = f[26]; p.b_ff1 = f[27]; p.w_ff2 = f[28]; p.b_ff2 = f[29];
  p.ln2_g = f[30]; p.ln2_b = f[31];
  p.w_stats = f[32]; p.b_stats = f[33];
  p.w_seat = f[34];  p.b_seat = f[35];
  p.w_comb = f[36];  p.b_comb = f[37];

  float* wsf = (float*)d_ws;
  p.ws_hist = wsf;
  p.ws_set  = wsf + (size_t)BATCH * 64;
  _Float16* h16 = (_Float16*)(wsf + (size_t)BATCH * (64 + 224));
  p.hp = h16;        h16 += 64 * 64;
  p.ewin = h16;      h16 += 192 * 64;
  p.ewout = h16;     h16 += 64 * 64;
  p.ff1 = h16;       h16 += 128 * 64;
  p.ff2 = h16;       h16 += 64 * 128;
  p.cwin[0] = h16;   h16 += 96 * 32;
  p.cwout[0] = h16;  h16 += 32 * 32;
  p.cwin[1] = h16;   h16 += 96 * 32;
  p.cwout[1] = h16;  h16 += 32 * 32;
  p.mrg = h16;       h16 += 64 * 224;
  p.stt = h16;       h16 += 64 * 384;
  p.cmb = h16;       h16 += 256 * 224;
  p.out = (float*)d_out;

  prep_kernel<<<128, 256, 0, stream>>>(p);
  hist_kernel<<<BATCH, 256, 0, stream>>>(p);
  set_kernel<<<BATCH, 256, 0, stream>>>(p);
  tail_kernel<<<BATCH / 16, 256, 0, stream>>>(p);
}
